// A2M_54296976556423
// MI455X (gfx1250) — matrix operand order repaired and checked by reading, not compiled, not run on hardware
//
#include <hip/hip_runtime.h>
#include <hip/hip_bf16.h>
#include <stdint.h>

// ---------------------------------------------------------------------------
// A2M (LaneGCN actor->map attention) for MI455X / gfx1250.
// - All GEMMs: v_wmma_f32_16x16x32_f16 (f16 inputs, f32 accumulate).
// - GroupNorm / ReLU / residual in f32 via LDS staging between WMMA stages.
// - LDS staging of raw rows uses GLOBAL_LOAD_ASYNC_TO_LDS_B128 (ASYNCcnt).
// - Scatter (index_add over hi) via f32 global atomics (a-buffer L2-resident).
// Workspace requirement: ~78 MB (f16 weights + f16 feat + f32 a-buffer).
// ---------------------------------------------------------------------------

typedef __attribute__((ext_vector_type(16))) _Float16 v16h;
typedef __attribute__((ext_vector_type(8)))  _Float16 v8h;
typedef __attribute__((ext_vector_type(8)))  float    v8f;

#define DDIM     128
#define TILE_M   64
#define NTHREADS 128
#define NMAP     100000
#define NACT     512

// ----------------------- CDNA5 async copy to LDS ---------------------------
// global_load_async_to_lds_b128: per-lane 16B global->LDS copy, no VGPR data,
// tracked with ASYNCcnt (ISA 10.7 / 15.18.3). LDS operand is the wave-relative
// LDS byte offset (= low 32 bits of a generic LDS pointer, per aperture rule).
__device__ __forceinline__ void async_ld_b128(void* lds, const void* g) {
  unsigned lo = (unsigned)(uintptr_t)lds;
  unsigned long long ga = (unsigned long long)(uintptr_t)g;
  asm volatile("global_load_async_to_lds_b128 %0, %1, off"
               :: "v"(lo), "v"(ga) : "memory");
}
__device__ __forceinline__ void async_wait0() {
  asm volatile("s_wait_asynccnt 0" ::: "memory");
}

__device__ __forceinline__ v8f wmma16(v16h a, v16h b, v8f c) {
  // D = A(16x32 f16) x B(32x16 f16) + C(16x16 f32)
  return __builtin_amdgcn_wmma_f32_16x16x32_f16(false, a, false, b, (short)0, c,
                                                false, false);
}

// A-matrix fragment (16x32 f16, M rows striped over lanes, interleaved K per
// ISA 7.12.2: VGPR0..3 hold K=kh..kh+7, VGPR4..7 hold K=16+kh..16+kh+7,
// kh = 0 (lanes 0-15) or 8 (lanes 16-31)).
__device__ __forceinline__ v16h load_frag_a(const _Float16* base, int lda,
                                            int m0, int k0, int lane) {
  const int m  = m0 + (lane & 15);
  const int kh = (lane >> 4) << 3;
  const _Float16* p = base + m * lda + k0 + kh;
  v8h lo = *(const v8h*)(p);
  v8h hi = *(const v8h*)(p + 16);
  v16h f;
#pragma unroll
  for (int i = 0; i < 8; ++i) { f[i] = lo[i]; f[i + 8] = hi[i]; }
  return f;
}

// B-matrix fragment (32x16 f16): lane n holds K=kh..kh+7 and 16+kh..16+kh+7
// (kh = 0 lanes 0-15, 8 lanes 16-31) of column n.  W is [N,K] row-major so
// out = act @ W^T maps B[k][n] = W[n][k] with K-contiguous per-lane loads.
__device__ __forceinline__ v16h load_frag_b(const _Float16* W, int ldk,
                                            int n0, int k0, int lane) {
  const int n  = n0 + (lane & 15);
  const int ks = k0 + ((lane >> 4) << 3);
  const _Float16* p = W + n * ldk + ks;
  v8h lo = *(const v8h*)(p);
  v8h hi = *(const v8h*)(p + 16);
  v16h f;
#pragma unroll
  for (int i = 0; i < 8; ++i) { f[i] = lo[i]; f[i + 8] = hi[i]; }
  return f;
}

// One wave computes a 16x128 f32 tile: rows [m0,m0+16) of A (LDS f16) times
// W^T (global f16, [128,K] row-major).
template <int K>
__device__ __forceinline__ void gemm16x128(const _Float16* A, int lda, int m0,
                                           const _Float16* W, int ldk, int lane,
                                           v8f acc[8]) {
#pragma unroll
  for (int kt = 0; kt < K; kt += 32) {
    v16h a = load_frag_a(A, lda, m0, kt, lane);
#pragma unroll
    for (int nt = 0; nt < 8; ++nt) {
      v16h b = load_frag_b(W, ldk, nt * 16, kt, lane);
      acc[nt] = wmma16(a, b, acc[nt]);
    }
  }
}

// C/D layout (ISA 7.12.2): VGPR r, lanes 0-15 -> M=r, N=lane; lanes 16-31 ->
// M=8+r, N=lane-16.
__device__ __forceinline__ void store_acc_lds(float* S, int m0, int lane,
                                              const v8f acc[8]) {
  const int n  = lane & 15;
  const int mo = m0 + ((lane >> 4) << 3);
#pragma unroll
  for (int nt = 0; nt < 8; ++nt)
#pragma unroll
    for (int r = 0; r < 8; ++r)
      S[(mo + r) * DDIM + nt * 16 + n] = acc[nt][r];
}

// Per-row GroupNorm(groups=1) over 128 f32 values in LDS -> f16 destination.
__device__ __forceinline__ void gn_apply_f16(const float* x, _Float16* dst,
                                             const float* gw, const float* gb,
                                             bool relu) {
  float s = 0.f, s2 = 0.f;
  for (int c = 0; c < DDIM; ++c) { float v = x[c]; s += v; s2 += v * v; }
  const float m   = s * (1.f / DDIM);
  const float var = s2 * (1.f / DDIM) - m * m;
  const float inv = rsqrtf(var + 1e-5f);
  for (int c = 0; c < DDIM; ++c) {
    float y = (x[c] - m) * inv * gw[c] + gb[c];
    if (relu) y = fmaxf(y, 0.f);
    dst[c] = (_Float16)y;
  }
}

// ------------------------------ small helpers ------------------------------

__global__ void k_f32_to_f16(const float* __restrict__ s,
                             _Float16* __restrict__ d, int n) {
  for (int i = blockIdx.x * blockDim.x + threadIdx.x; i < n;
       i += gridDim.x * blockDim.x)
    d[i] = (_Float16)s[i];
}

// meta_w is [128,130]; pad each row to ldk=136 halves so 16B fragment loads
// stay aligned (the 2 pris columns are applied in f32 by VALU in k_meta).
__global__ void k_meta_pad(const float* __restrict__ s,
                           _Float16* __restrict__ d) {
  int i = blockIdx.x * blockDim.x + threadIdx.x;
  if (i >= 128 * 136) return;
  int n = i / 136, k = i % 136;
  d[i] = (k < 130) ? (_Float16)s[n * 130 + k] : (_Float16)0.f;
}

// ------------------------------- meta layer --------------------------------
// feat = relu(GN(cat(feat, pris) @ meta_w^T))
__global__ __launch_bounds__(NTHREADS) void k_meta(
    const float* __restrict__ feat, const float* __restrict__ pris,
    const float* __restrict__ metaW, const _Float16* __restrict__ metaW16,
    const float* __restrict__ gw, const float* __restrict__ gb,
    float* __restrict__ out) {
  extern __shared__ char smem[];
  float*    S = (float*)smem;                    // 64*128*4 = 32768
  float*    P = (float*)(smem + 32768);          // 64*2*4   = 512
  _Float16* A = (_Float16*)(smem + 33280);       // 64*128*2 = 16384
  const int t = threadIdx.x, lane = t & 31, wv = t >> 5;
  const int r0 = blockIdx.x * TILE_M;
  for (int i = t; i < TILE_M * DDIM; i += NTHREADS) {  // f32->f16 convert
    int r = i >> 7, c = i & 127;
    int g = r0 + r; if (g > NMAP - 1) g = NMAP - 1;
    A[i] = (_Float16)feat[g * DDIM + c];
  }
  { int r = t >> 1, c = t & 1;
    int g = r0 + r; if (g > NMAP - 1) g = NMAP - 1;
    P[t] = pris[g * 2 + c]; }
  __syncthreads();
  const v8f vz = {0.f,0.f,0.f,0.f,0.f,0.f,0.f,0.f};
  v8f acc[8];
#pragma unroll
  for (int i = 0; i < 8; ++i) acc[i] = vz;
  gemm16x128<128>(A, DDIM, wv * 16, metaW16, 136, lane, acc);
  {  // pris tail columns (K=128,129) in f32
    const int nl = lane & 15;
    const int mo = wv * 16 + ((lane >> 4) << 3);
#pragma unroll
    for (int nt = 0; nt < 8; ++nt) {
      int n = nt * 16 + nl;
      float w0 = metaW[n * 130 + 128], w1 = metaW[n * 130 + 129];
#pragma unroll
      for (int r = 0; r < 8; ++r) {
        int m = mo + r;
        acc[nt][r] += P[m * 2 + 0] * w0 + P[m * 2 + 1] * w1;
      }
    }
  }
  store_acc_lds(S, wv * 16, lane, acc);
  __syncthreads();
  for (int r = t; r < TILE_M; r += NTHREADS) {
    int g = r0 + r; if (g >= NMAP) continue;
    const float* x = S + r * DDIM;
    float s = 0.f, s2 = 0.f;
    for (int c = 0; c < DDIM; ++c) { float v = x[c]; s += v; s2 += v * v; }
    float m = s * (1.f/DDIM), var = s2 * (1.f/DDIM) - m*m;
    float inv = rsqrtf(var + 1e-5f);
    for (int c = 0; c < DDIM; ++c) {
      float y = (x[c] - m) * inv * gw[c] + gb[c];
      out[g * DDIM + c] = fmaxf(y, 0.f);
    }
  }
}

// ----------------------------- a = feat @ agt^T ----------------------------
__global__ __launch_bounds__(NTHREADS) void k_agt(
    const _Float16* __restrict__ feat16, const _Float16* __restrict__ W,
    float* __restrict__ a) {
  __shared__ _Float16 A[TILE_M * DDIM];
  const int t = threadIdx.x, lane = t & 31, wv = t >> 5;
  const int r0 = blockIdx.x * TILE_M;
  // async global->LDS stage: 64 rows x 16 chunks of 16B
  for (int i = t; i < TILE_M * 16; i += NTHREADS) {
    int r = i >> 4, c = i & 15;
    int g = r0 + r; if (g > NMAP - 1) g = NMAP - 1;
    async_ld_b128(A + r * DDIM + c * 8, feat16 + g * DDIM + c * 8);
  }
  async_wait0();
  __syncthreads();
  const v8f vz = {0.f,0.f,0.f,0.f,0.f,0.f,0.f,0.f};
  v8f acc[8];
#pragma unroll
  for (int i = 0; i < 8; ++i) acc[i] = vz;
  gemm16x128<128>(A, DDIM, wv * 16, W, DDIM, lane, acc);
  const int nl = lane & 15;
  const int mo = wv * 16 + ((lane >> 4) << 3);
#pragma unroll
  for (int nt = 0; nt < 8; ++nt)
#pragma unroll
    for (int r = 0; r < 8; ++r) {
      int g = r0 + mo + r;
      if (g < NMAP) a[g * DDIM + nt * 16 + nl] = acc[nt][r];
    }
}

// ------------------------- fused per-edge attention ------------------------
__global__ __launch_bounds__(NTHREADS) void k_edge(
    const _Float16* __restrict__ feat16, const _Float16* __restrict__ actors16,
    const float* __restrict__ map_ctrs, const float* __restrict__ actor_ctrs,
    const int* __restrict__ hi, const int* __restrict__ wi, int E,
    const float* __restrict__ dW0, const float* __restrict__ dB0,
    const _Float16* __restrict__ dW1, const float* __restrict__ d_gw,
    const float* __restrict__ d_gb,
    const _Float16* __restrict__ qW, const float* __restrict__ q_gw,
    const float* __restrict__ q_gb,
    const _Float16* __restrict__ cW0, const float* __restrict__ c_gw,
    const float* __restrict__ c_gb,
    const _Float16* __restrict__ cW1, float* __restrict__ a) {
  extern __shared__ char smem[];
  float*    S   = (float*)smem;                        // 32768
  float*    REL = (float*)(smem + 32768);              // 512
  int*      HI  = (int*)(smem + 33280);                // 256
  int*      WI  = (int*)(smem + 33536);                // 256
  _Float16* C3  = (_Float16*)(smem + 33792);           // 64*384*2 = 49152
  _Float16* A1  = (_Float16*)(smem + 33792 + 49152);   // 64*128*2 = 16384
  const int t = threadIdx.x, lane = t & 31, wv = t >> 5;
  const int e0 = blockIdx.x * TILE_M;
  if (t < TILE_M) {
    int e = e0 + t;
    bool v = e < E;
    int h = v ? hi[e] : 0;
    int w = v ? wi[e] : 0;
    HI[t] = h; WI[t] = w;
    REL[t * 2 + 0] = map_ctrs[h * 2 + 0] - actor_ctrs[w * 2 + 0];
    REL[t * 2 + 1] = map_ctrs[h * 2 + 1] - actor_ctrs[w * 2 + 1];
  }
  __syncthreads();
  // async gathered-row copies: actors[wi] -> C3[:,256:384], feat16[hi] -> A1
  for (int i = t; i < TILE_M * 16; i += NTHREADS) {
    int r = i >> 4, c = i & 15;
    async_ld_b128(C3 + r * 384 + 256 + c * 8, actors16 + WI[r] * DDIM + c * 8);
    async_ld_b128(A1 + r * DDIM + c * 8, feat16 + HI[r] * DDIM + c * 8);
  }
  async_wait0();
  __syncthreads();
  const v8f vz = {0.f,0.f,0.f,0.f,0.f,0.f,0.f,0.f};
  v8f acc[8];
  // q = relu(GN(feat[hi] @ query^T)) -> C3[:,128:256]
#pragma unroll
  for (int i = 0; i < 8; ++i) acc[i] = vz;
  gemm16x128<128>(A1, DDIM, wv * 16, qW, DDIM, lane, acc);
  store_acc_lds(S, wv * 16, lane, acc);
  __syncthreads();
  for (int r = t; r < TILE_M; r += NTHREADS)
    gn_apply_f16(S + r * DDIM, C3 + r * 384 + 128, q_gw, q_gb, true);
  __syncthreads();
  // d0 = relu(rel @ dist_w0^T + b0), K=2: pure VALU -> A1
  for (int i = t; i < TILE_M * DDIM; i += NTHREADS) {
    int r = i >> 7, c = i & 127;
    float y = REL[r * 2 + 0] * dW0[c * 2 + 0] +
              REL[r * 2 + 1] * dW0[c * 2 + 1] + dB0[c];
    A1[i] = (_Float16)fmaxf(y, 0.f);
  }
  __syncthreads();
  // d = relu(GN(d0 @ dist_w1^T)) -> C3[:,0:128]
#pragma unroll
  for (int i = 0; i < 8; ++i) acc[i] = vz;
  gemm16x128<128>(A1, DDIM, wv * 16, dW1, DDIM, lane, acc);
  store_acc_lds(S, wv * 16, lane, acc);
  __syncthreads();
  for (int r = t; r < TILE_M; r += NTHREADS)
    gn_apply_f16(S + r * DDIM, C3 + r * 384, d_gw, d_gb, true);
  __syncthreads();
  // h = relu(GN(C3 @ ctx_w0^T)), K=384 -> A1
#pragma unroll
  for (int i = 0; i < 8; ++i) acc[i] = vz;
  gemm16x128<384>(C3, 384, wv * 16, cW0, 384, lane, acc);
  store_acc_lds(S, wv * 16, lane, acc);
  __syncthreads();
  for (int r = t; r < TILE_M; r += NTHREADS)
    gn_apply_f16(S + r * DDIM, A1 + r * DDIM, c_gw, c_gb, true);
  __syncthreads();
  // c = h @ ctx_w1^T ; scatter: a[hi] += c
#pragma unroll
  for (int i = 0; i < 8; ++i) acc[i] = vz;
  gemm16x128<128>(A1, DDIM, wv * 16, cW1, DDIM, lane, acc);
  const int nl = lane & 15;
  const int mo = wv * 16 + ((lane >> 4) << 3);
#pragma unroll
  for (int nt = 0; nt < 8; ++nt)
#pragma unroll
    for (int r = 0; r < 8; ++r) {
      int m = mo + r;
      if (e0 + m < E)
        atomicAdd(&a[HI[m] * DDIM + nt * 16 + nl], acc[nt][r]);
    }
}

// -------------------- a -> GN,relu -> lin -> GN -> +res,relu ---------------
__global__ __launch_bounds__(NTHREADS) void k_node2(
    const float* __restrict__ a, const float* __restrict__ n_gw,
    const float* __restrict__ n_gb, const _Float16* __restrict__ lW,
    const float* __restrict__ l_gw, const float* __restrict__ l_gb,
    float* __restrict__ feat) {
  extern __shared__ char smem[];
  float*    S  = (float*)smem;                 // 32768
  _Float16* A1 = (_Float16*)(smem + 32768);    // 16384
  const int t = threadIdx.x, lane = t & 31, wv = t >> 5;
  const int r0 = blockIdx.x * TILE_M;
  // async global->LDS stage of f32 a-rows: 64 rows x 32 chunks of 16B
  for (int i = t; i < TILE_M * 32; i += NTHREADS) {
    int r = i >> 5, c = i & 31;
    int g = r0 + r; if (g > NMAP - 1) g = NMAP - 1;
    async_ld_b128(S + r * DDIM + c * 4, a + g * DDIM + c * 4);
  }
  async_wait0();
  __syncthreads();
  for (int r = t; r < TILE_M; r += NTHREADS)
    gn_apply_f16(S + r * DDIM, A1 + r * DDIM, n_gw, n_gb, true);
  __syncthreads();
  const v8f vz = {0.f,0.f,0.f,0.f,0.f,0.f,0.f,0.f};
  v8f acc[8];
#pragma unroll
  for (int i = 0; i < 8; ++i) acc[i] = vz;
  gemm16x128<128>(A1, DDIM, wv * 16, lW, DDIM, lane, acc);
  store_acc_lds(S, wv * 16, lane, acc);
  __syncthreads();
  for (int r = t; r < TILE_M; r += NTHREADS) {
    int g = r0 + r; if (g >= NMAP) continue;
    const float* x = S + r * DDIM;
    float s = 0.f, s2 = 0.f;
    for (int c = 0; c < DDIM; ++c) { float v = x[c]; s += v; s2 += v * v; }
    float m = s * (1.f/DDIM), var = s2 * (1.f/DDIM) - m * m;
    float inv = rsqrtf(var + 1e-5f);
    for (int c = 0; c < DDIM; ++c) {
      float y = (x[c] - m) * inv * l_gw[c] + l_gb[c];
      feat[g * DDIM + c] = fmaxf(y + feat[g * DDIM + c], 0.f);
    }
  }
}

// --------------------------------- launch ----------------------------------
extern "C" void kernel_launch(void* const* d_in, const int* in_sizes, int n_in,
                              void* d_out, int out_size, void* d_ws,
                              size_t ws_size, hipStream_t stream) {
  const float* feat       = (const float*)d_in[0];
  const float* pris       = (const float*)d_in[1];
  const float* map_ctrs   = (const float*)d_in[2];
  const float* actors     = (const float*)d_in[3];
  const float* actor_ctrs = (const float*)d_in[4];
  const float* meta_w     = (const float*)d_in[5];
  const float* meta_gw    = (const float*)d_in[6];
  const float* meta_gb    = (const float*)d_in[7];
  const float* dist_w0    = (const float*)d_in[8];
  const float* dist_b0    = (const float*)d_in[9];
  const float* dist_w1    = (const float*)d_in[10];
  const float* dist_gw    = (const float*)d_in[11];
  const float* dist_gb    = (const float*)d_in[12];
  const float* query_w    = (const float*)d_in[13];
  const float* query_gw   = (const float*)d_in[14];
  const float* query_gb   = (const float*)d_in[15];
  const float* ctx_w0     = (const float*)d_in[16];
  const float* ctx_gw0    = (const float*)d_in[17];
  const float* ctx_gb0    = (const float*)d_in[18];
  const float* ctx_w1     = (const float*)d_in[19];
  const float* agt_w      = (const float*)d_in[20];
  const float* norm_gw    = (const float*)d_in[21];
  const float* norm_gb    = (const float*)d_in[22];
  const float* lin_w      = (const float*)d_in[23];
  const float* lin_gw     = (const float*)d_in[24];
  const float* lin_gb     = (const float*)d_in[25];
  const int*   hi         = (const int*)d_in[26];
  const int*   wi         = (const int*)d_in[27];
  const int E = in_sizes[26];
  float* out = (float*)d_out;

  // workspace carve (f16 weights | f16 feat | f32 a-buffer) ~78 MB total
  char* ws = (char*)d_ws;
  _Float16* actors16  = (_Float16*)ws;           // 512*128
  _Float16* metaW16   = actors16 + 65536;        // 128*136
  _Float16* distW1_16 = metaW16 + 17408;         // 2*128*128
  _Float16* queryW16  = distW1_16 + 32768;
  _Float16* ctxW0_16  = queryW16 + 32768;        // 2*128*384
  _Float16* ctxW1_16  = ctxW0_16 + 98304;
  _Float16* agtW16    = ctxW1_16 + 32768;
  _Float16* linW16    = agtW16 + 32768;
  _Float16* feat16    = (_Float16*)(ws + 690176);        // 100000*128 f16
  float*    abuf      = (float*)(ws + 26290176);         // 100000*128 f32

  auto conv = [&](const float* s, _Float16* d, int n) {
    k_f32_to_f16<<<(n + 255) / 256, 256, 0, stream>>>(s, d, n);
  };
  conv(actors, actors16, NACT * DDIM);
  conv(dist_w1, distW1_16, 2 * 128 * 128);
  conv(query_w, queryW16, 2 * 128 * 128);
  conv(ctx_w0, ctxW0_16, 2 * 128 * 384);
  conv(ctx_w1, ctxW1_16, 2 * 128 * 128);
  conv(agt_w, agtW16, 2 * 128 * 128);
  conv(lin_w, linW16, 2 * 128 * 128);
  k_meta_pad<<<(128 * 136 + 255) / 256, 256, 0, stream>>>(meta_w, metaW16);

  const int gridMap = (NMAP + TILE_M - 1) / TILE_M;
  const int gridE   = (E + TILE_M - 1) / TILE_M;
  const size_t META_SMEM  = 32768 + 512 + 16384;
  const size_t EDGE_SMEM  = 32768 + 512 + 256 + 256 + 49152 + 16384;
  const size_t NODE2_SMEM = 32768 + 16384;

  k_meta<<<gridMap, NTHREADS, META_SMEM, stream>>>(feat, pris, meta_w, metaW16,
                                                   meta_gw, meta_gb, out);
  for (int l = 0; l < 2; ++l) {
    k_f32_to_f16<<<(NMAP * DDIM + 255) / 256, 256, 0, stream>>>(out, feat16,
                                                                NMAP * DDIM);
    k_agt<<<gridMap, NTHREADS, 0, stream>>>(feat16, agtW16 + l * 16384, abuf);
    k_edge<<<gridE, NTHREADS, EDGE_SMEM, stream>>>(
        feat16, actors16, map_ctrs, actor_ctrs, hi, wi, E,
        dist_w0 + l * 256, dist_b0 + l * 128,
        distW1_16 + l * 16384, dist_gw + l * 128, dist_gb + l * 128,
        queryW16 + l * 16384, query_gw + l * 128, query_gb + l * 128,
        ctxW0_16 + l * 49152, ctx_gw0 + l * 128, ctx_gb0 + l * 128,
        ctxW1_16 + l * 16384, abuf);
    k_node2<<<gridMap, NTHREADS, NODE2_SMEM, stream>>>(
        abuf, norm_gw + l * 128, norm_gb + l * 128, linW16 + l * 16384,
        lin_gw + l * 128, lin_gb + l * 128, out);
  }
  (void)n_in; (void)out_size; (void)ws_size;
}
